// DDI_Encoder_15367392985220
// MI455X (gfx1250) — matrix operand order repaired and checked by reading, not compiled, not run on hardware
//
#include <hip/hip_runtime.h>

// ---------------------------------------------------------------------------
// RGCN (DDI encoder) on gfx1250 using v_wmma_f32_16x16x32_bf16.
//
//   prep : drug_w -> bf16 transposed; W1t/W2t = comp x basis (bf16, o-major);
//          R1t/R2t = root transposed bf16 (layer2 padded to 32 cols)
//   deg  : deg[r][d] = #edges (r,*,d) via f32 atomics
//   g1   : x_bf = bf16(x_drug @ drug_w)     (WMMA bf16, M=10000 K=2048 N=128)
//   g2   : h    = x_bf @ root1 + bias1      (WMMA bf16, K=128 N=64, f32 out)
//   m1   : h   += scatter_r,e msg1/deg      (WMMA bf16 + atomic f32 adds)
//   relu : h_bf = bf16(max(h,0))
//   g3   : out  = h_bf @ root2 + bias2      (WMMA bf16, K=64 N=20 pad 32)
//   m2   : out += scatter_r,e msg2/deg
//
// x / h are only ever consumed as WMMA bf16 operands, so they are stored in
// bf16 row-major; A fragments then load as two contiguous b128s per tile with
// no per-read conversion VALU.
// ---------------------------------------------------------------------------

typedef __attribute__((ext_vector_type(16))) __bf16       v16bf;
typedef __attribute__((ext_vector_type(8)))  float        v8f;
typedef __attribute__((ext_vector_type(8)))  unsigned int v8u;

__device__ __forceinline__ unsigned short f2bf(float f) {
  unsigned a = __float_as_uint(f);
  return (unsigned short)((a + 0x7fffu + ((a >> 16) & 1u)) >> 16);
}

__device__ __forceinline__ unsigned pack2bf(float lo, float hi) {
  unsigned a = __float_as_uint(lo);
  unsigned b = __float_as_uint(hi);
  a = (a + 0x7fffu + ((a >> 16) & 1u)) >> 16;
  b = (b + 0x7fffu + ((b >> 16) & 1u)) & 0xffff0000u;
  return (a & 0xffffu) | b;
}

__device__ __forceinline__ void atomAddF(float* p, float v) {
  __hip_atomic_fetch_add(p, v, __ATOMIC_RELAXED, __HIP_MEMORY_SCOPE_AGENT);
}

// ---------------- prep kernels -------------------------------------------

// Wt[c*2048+k] = bf16(drug_w[k*128+c]) ; 128x2048
__global__ void k_conv_wt(const float* __restrict__ w, unsigned short* __restrict__ wt) {
  int tid = blockIdx.x * blockDim.x + threadIdx.x;
  if (tid >= 128 * 2048) return;
  int c = tid / 2048, k = tid % 2048;
  wt[tid] = f2bf(w[k * 128 + c]);
}

// wt[r][o][i] = bf16(sum_b comp[r,b]*basis[b,i,o]), o >= Ovalid -> 0
__global__ void k_basis(const float* __restrict__ comp, const float* __restrict__ basis,
                        unsigned short* __restrict__ wt, int I, int Ovalid, int Opad) {
  int tid = blockIdx.x * blockDim.x + threadIdx.x;
  int total = 100 * Opad * I;
  if (tid >= total) return;
  int r = tid / (Opad * I);
  int rem = tid % (Opad * I);
  int o = rem / I, i = rem % I;
  float acc = 0.f;
  if (o < Ovalid) {
#pragma unroll
    for (int b = 0; b < 8; ++b)
      acc += comp[r * 8 + b] * basis[(b * I + i) * Ovalid + o];
  }
  wt[tid] = f2bf(acc);
}

// rt[o][i] = bf16(root[i,o]), o >= Ovalid -> 0
__global__ void k_root(const float* __restrict__ root, unsigned short* __restrict__ rt,
                       int I, int Ovalid, int Opad) {
  int tid = blockIdx.x * blockDim.x + threadIdx.x;
  if (tid >= Opad * I) return;
  int o = tid / I, i = tid % I;
  rt[tid] = f2bf((o < Ovalid) ? root[i * Ovalid + o] : 0.f);
}

__global__ void k_zero(float* __restrict__ p, int n) {
  int tid = blockIdx.x * blockDim.x + threadIdx.x;
  if (tid < n) p[tid] = 0.f;
}

__global__ void k_deg(const int* __restrict__ ei, float* __restrict__ deg) {
  int tid = blockIdx.x * blockDim.x + threadIdx.x;
  if (tid >= 100 * 5000) return;
  int r = tid / 5000, e = tid % 5000;
  int d = ei[r * 10000 + 5000 + e];
  atomAddF(&deg[r * 10000 + d], 1.0f);
}

// h_bf = bf16(relu(h))
__global__ void k_relu_cvt(const float* __restrict__ h, unsigned short* __restrict__ hb,
                           int n) {
  int tid = blockIdx.x * blockDim.x + threadIdx.x;
  if (tid < n) hb[tid] = f2bf(fmaxf(h[tid], 0.f));
}

// ---------------- WMMA fragment helpers ----------------------------------

// A fragment from f32 source (inline conversion): row = lane&15.
template <int K>
__device__ __forceinline__ v16bf loadAfragF32(const float* __restrict__ A, size_t aBase,
                                              int k0, int khA) {
  v8u au;
#pragma unroll
  for (int v = 0; v < 8; ++v) {
    int kb = k0 + ((v & 4) << 2) + khA + ((v & 3) << 1);
    float2 f = *reinterpret_cast<const float2*>(A + aBase + kb);
    au[v] = pack2bf(f.x, f.y);
  }
  return __builtin_bit_cast(v16bf, au);
}

// A fragment from bf16 source (as uint pairs): two contiguous 4-dword groups.
__device__ __forceinline__ v16bf loadAfragBF(const unsigned* __restrict__ Au,
                                             size_t rowBase, int k0, int khA) {
  v8u au;
  size_t b = rowBase + ((k0 + khA) >> 1);
#pragma unroll
  for (int v = 0; v < 4; ++v) au[v] = Au[b + v];
#pragma unroll
  for (int v = 0; v < 4; ++v) au[4 + v] = Au[b + 8 + v];
  return __builtin_bit_cast(v16bf, au);
}

// B fragment: 32x16 bf16, col = lane&15; Bt is (N x K) bf16 as uint pairs.
__device__ __forceinline__ v16bf loadBfrag(const unsigned* __restrict__ BtU,
                                           size_t colBase, int k0, int khB) {
  v8u bu;
#pragma unroll
  for (int v = 0; v < 8; ++v) {
    int kb = k0 + khB + ((v & 4) << 2) + ((v & 3) << 1);
    bu[v] = BtU[colBase + (kb >> 1)];
  }
  return __builtin_bit_cast(v16bf, bu);
}

// ---------- GEMM 1: C_bf16 = bf16( A_f32(MxK) @ Bt^T ) -------------------

template <int KT, int NT>
__global__ void wmma_gemm_f2b(const float* __restrict__ A, const unsigned* __restrict__ BtU,
                              unsigned short* __restrict__ Cb, int mTiles) {
  constexpr int K = KT * 32;
  constexpr int Nout = NT * 16;
  int w = blockIdx.x * (blockDim.x >> 5) + (threadIdx.x >> 5);
  if (w >= mTiles) return;
  int lane = threadIdx.x & 31;
  int col = lane & 15;
  int khA = (lane & 16) ? 8 : 0;
  int khB = (lane & 16) ? 8 : 0;
  size_t aBase = (size_t)(w * 16 + col) * K;

  v8f zero = {};
  v8f c[NT];
#pragma unroll
  for (int n = 0; n < NT; ++n) c[n] = zero;

  for (int kt = 0; kt < KT; ++kt) {
    int k0 = kt * 32;
    v16bf a = loadAfragF32<K>(A, aBase, k0, khA);
#pragma unroll
    for (int n = 0; n < NT; ++n) {
      size_t colBase = (size_t)(n * 16 + col) * (K >> 1);
      v16bf b = loadBfrag(BtU, colBase, k0, khB);
      c[n] = __builtin_amdgcn_wmma_f32_16x16x32_bf16(false, a, false, b, (short)0,
                                                     c[n], false, false);
    }
  }

  int rowOff = (lane & 16) ? 8 : 0;
#pragma unroll
  for (int v = 0; v < 8; ++v) {
    size_t r = (size_t)(w * 16 + v + rowOff);
#pragma unroll
    for (int n = 0; n < NT; ++n) Cb[r * Nout + n * 16 + col] = f2bf(c[n][v]);
  }
}

// ---------- GEMM 2/3: C_f32 = A_bf16(MxK) @ Bt^T + bias ------------------

template <int KT, int NT>
__global__ void wmma_gemm_b2f(const unsigned* __restrict__ Au, const unsigned* __restrict__ BtU,
                              const float* __restrict__ bias, float* __restrict__ C,
                              int mTiles, int Nvalid) {
  constexpr int K = KT * 32;
  int w = blockIdx.x * (blockDim.x >> 5) + (threadIdx.x >> 5);
  if (w >= mTiles) return;
  int lane = threadIdx.x & 31;
  int col = lane & 15;
  int khA = (lane & 16) ? 8 : 0;
  int khB = (lane & 16) ? 8 : 0;
  size_t aBase = (size_t)(w * 16 + col) * (K >> 1);

  v8f zero = {};
  v8f c[NT];
#pragma unroll
  for (int n = 0; n < NT; ++n) c[n] = zero;

#pragma unroll
  for (int kt = 0; kt < KT; ++kt) {
    int k0 = kt * 32;
    v16bf a = loadAfragBF(Au, aBase, k0, khA);
#pragma unroll
    for (int n = 0; n < NT; ++n) {
      size_t colBase = (size_t)(n * 16 + col) * (K >> 1);
      v16bf b = loadBfrag(BtU, colBase, k0, khB);
      c[n] = __builtin_amdgcn_wmma_f32_16x16x32_bf16(false, a, false, b, (short)0,
                                                     c[n], false, false);
    }
  }

  int rowOff = (lane & 16) ? 8 : 0;
#pragma unroll
  for (int v = 0; v < 8; ++v) {
    size_t r = (size_t)(w * 16 + v + rowOff);
#pragma unroll
    for (int n = 0; n < NT; ++n) {
      int gc = n * 16 + col;
      if (gc < Nvalid) C[r * Nvalid + gc] = c[n][v] + bias[gc];
    }
  }
}

// ------------- per-relation message GEMM + normalized scatter ------------

template <int KT, int NT>
__global__ void wmma_msg(const unsigned* __restrict__ Xu, const unsigned* __restrict__ WtU,
                         const int* __restrict__ ei, const float* __restrict__ deg,
                         float* __restrict__ out, int Nvalid) {
  constexpr int K = KT * 32;
  constexpr int E = 5000;
  constexpr int TPR = 313;  // ceil(5000/16)
  int w = blockIdx.x * (blockDim.x >> 5) + (threadIdx.x >> 5);
  if (w >= 100 * TPR) return;
  int r = w / TPR, t = w % TPR;
  int lane = threadIdx.x & 31;
  int col = lane & 15;
  int khA = (lane & 16) ? 8 : 0;
  int khB = (lane & 16) ? 8 : 0;

  int eA = t * 16 + col;
  int srcIdx = (eA < E) ? ei[r * 10000 + eA] : 0;  // padded rows -> row 0 (masked at store)
  size_t aBase = (size_t)srcIdx * (K >> 1);
  const unsigned* BtU = WtU + (size_t)r * (NT * 16) * (K >> 1);

  v8f zero = {};
  v8f c[NT];
#pragma unroll
  for (int n = 0; n < NT; ++n) c[n] = zero;

#pragma unroll
  for (int kt = 0; kt < KT; ++kt) {
    int k0 = kt * 32;
    v16bf a = loadAfragBF(Xu, aBase, k0, khA);
#pragma unroll
    for (int n = 0; n < NT; ++n) {
      size_t colBase = (size_t)(n * 16 + col) * (K >> 1);
      v16bf b = loadBfrag(BtU, colBase, k0, khB);
      c[n] = __builtin_amdgcn_wmma_f32_16x16x32_bf16(false, a, false, b, (short)0,
                                                     c[n], false, false);
    }
  }

  int rowOff = (lane & 16) ? 8 : 0;
#pragma unroll
  for (int v = 0; v < 8; ++v) {
    int e = t * 16 + v + rowOff;
    if (e < E) {
      int d = ei[r * 10000 + 5000 + e];
      float invd = 1.0f / fmaxf(deg[r * 10000 + d], 1.0f);
#pragma unroll
      for (int n = 0; n < NT; ++n) {
        int gc = n * 16 + col;
        if (gc < Nvalid) atomAddF(&out[(size_t)d * Nvalid + gc], c[n][v] * invd);
      }
    }
  }
}

// ---------------------------------------------------------------------------

extern "C" void kernel_launch(void* const* d_in, const int* in_sizes, int n_in,
                              void* d_out, int out_size, void* d_ws, size_t ws_size,
                              hipStream_t stream) {
  (void)in_sizes; (void)n_in; (void)out_size; (void)ws_size;
  const float* x_drug = (const float*)d_in[0];   // (10000,2048)
  const float* drug_w = (const float*)d_in[1];   // (2048,128)
  const int*   ei     = (const int*)d_in[2];     // (100,2,5000)
  const float* basis1 = (const float*)d_in[3];   // (8,128,64)
  const float* comp1  = (const float*)d_in[4];   // (100,8)
  const float* root1  = (const float*)d_in[5];   // (128,64)
  const float* bias1  = (const float*)d_in[6];   // (64,)
  const float* basis2 = (const float*)d_in[7];   // (8,64,20)
  const float* comp2  = (const float*)d_in[8];   // (100,8)
  const float* root2  = (const float*)d_in[9];   // (64,20)
  const float* bias2  = (const float*)d_in[10];  // (20,)
  float* out = (float*)d_out;                    // (10000,20)

  char* ws = (char*)d_ws;
  unsigned short* x_bf = (unsigned short*)(ws + 0);        // 10000*128 bf16 (2,560,000 B)
  unsigned short* Wt   = (unsigned short*)(ws + 2560000);  // 128*2048 bf16  (524,288 B)
  unsigned short* W1t  = (unsigned short*)(ws + 3084288);  // 100*64*128 bf16(1,638,400 B)
  unsigned short* W2t  = (unsigned short*)(ws + 4722688);  // 100*32*64 bf16 (409,600 B)
  unsigned short* R1t  = (unsigned short*)(ws + 5132288);  // 64*128 bf16    (16,384 B)
  unsigned short* R2t  = (unsigned short*)(ws + 5148672);  // 32*64 bf16     (4,096 B)
  float*          deg  = (float*)(ws + 5152768);           // 100*10000 f32  (4,000,000 B)
  float*          h    = (float*)(ws + 9152768);           // 10000*64 f32   (2,560,000 B)
  unsigned short* h_bf = (unsigned short*)(ws + 11712768); // 10000*64 bf16  (1,280,000 B)

  dim3 B(256);
  // --- prep ---
  k_conv_wt<<<(128 * 2048 + 255) / 256, B, 0, stream>>>(drug_w, Wt);
  k_basis<<<(100 * 64 * 128 + 255) / 256, B, 0, stream>>>(comp1, basis1, W1t, 128, 64, 64);
  k_basis<<<(100 * 32 * 64 + 255) / 256, B, 0, stream>>>(comp2, basis2, W2t, 64, 20, 32);
  k_root<<<(64 * 128 + 255) / 256, B, 0, stream>>>(root1, R1t, 128, 64, 64);
  k_root<<<(32 * 64 + 255) / 256, B, 0, stream>>>(root2, R2t, 64, 20, 32);
  k_zero<<<(1000000 + 255) / 256, B, 0, stream>>>(deg, 1000000);
  k_deg<<<(500000 + 255) / 256, B, 0, stream>>>(ei, deg);

  // --- x_bf = bf16(x_drug @ drug_w)  (M=10000, K=2048, N=128) ---
  wmma_gemm_f2b<64, 8><<<79, B, 0, stream>>>(x_drug, (const unsigned*)Wt, x_bf, 625);

  // --- layer 1 ---
  wmma_gemm_b2f<4, 4><<<79, B, 0, stream>>>((const unsigned*)x_bf, (const unsigned*)R1t,
                                            bias1, h, 625, 64);
  wmma_msg<4, 4><<<3913, B, 0, stream>>>((const unsigned*)x_bf, (const unsigned*)W1t,
                                         ei, deg, h, 64);
  k_relu_cvt<<<(640000 + 255) / 256, B, 0, stream>>>(h, h_bf, 640000);

  // --- layer 2 ---
  wmma_gemm_b2f<2, 2><<<79, B, 0, stream>>>((const unsigned*)h_bf, (const unsigned*)R2t,
                                            bias2, out, 625, 20);
  wmma_msg<2, 2><<<3913, B, 0, stream>>>((const unsigned*)h_bf, (const unsigned*)W2t,
                                         ei, deg, out, 20);
}
